// MultiHeadAttention_55886114455776
// MI455X (gfx1250) — matrix operand order repaired and checked by reading, not compiled, not run on hardware
//
#include <hip/hip_runtime.h>
#include <hip/hip_bf16.h>
#include <stdint.h>

#define NUM_HEADS 32
#define NUM_KV_GROUPS 8
#define HEAD_DIM 64
#define GROUP_SIZE (NUM_HEADS / NUM_KV_GROUPS)

typedef __bf16 bf16_t;
typedef __attribute__((ext_vector_type(16))) __bf16 v16bf;
typedef __attribute__((ext_vector_type(8)))  __bf16 v8bf;
typedef __attribute__((ext_vector_type(8)))  float  v8f;
typedef __attribute__((ext_vector_type(4)))  float  v4f;

union Frag { v16bf v; v8bf h[2]; };

#define WMMA_BF16(A, B, C) \
  __builtin_amdgcn_wmma_f32_16x16x32_bf16(false, (A), false, (B), (short)0, (C), false, false)

__device__ __forceinline__ bf16_t f2bf(float f) {
  union { float f; uint32_t u; } in; in.f = f;
  uint32_t u = in.u;
  uint32_t r = u + 0x7FFFu + ((u >> 16) & 1u);   // round-to-nearest-even
  union { uint16_t s; bf16_t b; } out; out.s = (uint16_t)(r >> 16);
  return out.b;
}

// ---------------------------------------------------------------------------
// Vectorized f32 -> bf16 conversion (4 elems/thread); n must be multiple of 4.
// ---------------------------------------------------------------------------
__global__ void cvt_f32_bf16_x4(const float* __restrict__ in, bf16_t* __restrict__ out, int n4) {
  int i = blockIdx.x * blockDim.x + threadIdx.x;
  if (i >= n4) return;
  v4f x = *(const v4f*)(in + (size_t)i * 4);
  bf16_t* o = out + (size_t)i * 4;
  o[0] = f2bf(x.x); o[1] = f2bf(x.y); o[2] = f2bf(x.z); o[3] = f2bf(x.w);
}

// ---------------------------------------------------------------------------
// GEMM fragment set: one 32-K step (2 A-fragments for two 16-row tiles,
// 4 B-fragments for a 64-col tile).
// ---------------------------------------------------------------------------
struct FragSet { Frag a0, a1; v16bf b0, b1, b2, b3; };

__device__ __forceinline__ void load_frags(FragSet& f, int k, int hl,
                                           const bf16_t* __restrict__ a0p,
                                           const bf16_t* __restrict__ a1p,
                                           const bf16_t* __restrict__ w0,
                                           const bf16_t* __restrict__ w1,
                                           const bf16_t* __restrict__ w2,
                                           const bf16_t* __restrict__ w3) {
  f.a0.h[0] = *(const v8bf*)(a0p + k + hl * 16);
  f.a0.h[1] = *(const v8bf*)(a0p + k + 8 + hl * 16);
  f.a1.h[0] = *(const v8bf*)(a1p + k + hl * 16);
  f.a1.h[1] = *(const v8bf*)(a1p + k + 8 + hl * 16);
  f.b0 = *(const v16bf*)(w0 + k + hl * 16);
  f.b1 = *(const v16bf*)(w1 + k + hl * 16);
  f.b2 = *(const v16bf*)(w2 + k + hl * 16);
  f.b3 = *(const v16bf*)(w3 + k + hl * 16);
}

__device__ __forceinline__ void mma_step(v8f acc[2][4], const FragSet& f) {
  acc[0][0] = WMMA_BF16(f.a0.v, f.b0, acc[0][0]);
  acc[0][1] = WMMA_BF16(f.a0.v, f.b1, acc[0][1]);
  acc[0][2] = WMMA_BF16(f.a0.v, f.b2, acc[0][2]);
  acc[0][3] = WMMA_BF16(f.a0.v, f.b3, acc[0][3]);
  acc[1][0] = WMMA_BF16(f.a1.v, f.b0, acc[1][0]);
  acc[1][1] = WMMA_BF16(f.a1.v, f.b1, acc[1][1]);
  acc[1][2] = WMMA_BF16(f.a1.v, f.b2, acc[1][2]);
  acc[1][3] = WMMA_BF16(f.a1.v, f.b3, acc[1][3]);
}

// ---------------------------------------------------------------------------
// C[M][N] (f32) = A[M][K] (bf16, row-major) @ W[N][K]^T (bf16, row-major).
// grid = (N/512, M/32), block = 256 threads (8 waves).
// Each wave computes a 32(M) x 64(N) tile; K loop unrolled by 2 with
// ping-pong fragment buffers (no rotation copies, no spills —
// __launch_bounds__(256,1) relaxes the occupancy-driven VGPR cap).
// Requires K % 64 == 0 and K >= 64.
// ---------------------------------------------------------------------------
__global__ void __launch_bounds__(256, 1)
gemm_bf16_nt(const bf16_t* __restrict__ A, const bf16_t* __restrict__ W,
             float* __restrict__ C, int M, int N, int K) {
  const int lane = threadIdx.x & 31;
  const int wave = threadIdx.x >> 5;
  const int hl   = lane >> 4;     // 16-lane group (K-half selector)
  const int l16  = lane & 15;
  const int mt = blockIdx.y;              // 32-row tile
  const int nt = blockIdx.x * 8 + wave;   // 64-col tile

  const bf16_t* a0p = A + (size_t)(mt * 32 +  0 + l16) * K;
  const bf16_t* a1p = A + (size_t)(mt * 32 + 16 + l16) * K;
  const bf16_t* w0 = W + (size_t)(nt * 64 +  0 + l16) * K;
  const bf16_t* w1 = W + (size_t)(nt * 64 + 16 + l16) * K;
  const bf16_t* w2 = W + (size_t)(nt * 64 + 32 + l16) * K;
  const bf16_t* w3 = W + (size_t)(nt * 64 + 48 + l16) * K;

  v8f acc[2][4] = {};

  FragSet f0, f1;
  load_frags(f0, 0, hl, a0p, a1p, w0, w1, w2, w3);

  int k = 0;
  for (; k + 64 < K; k += 64) {
    load_frags(f1, k + 32, hl, a0p, a1p, w0, w1, w2, w3);
    mma_step(acc, f0);
    load_frags(f0, k + 64, hl, a0p, a1p, w0, w1, w2, w3);
    mma_step(acc, f1);
  }
  // tail: k == K - 64
  load_frags(f1, k + 32, hl, a0p, a1p, w0, w1, w2, w3);
  mma_step(acc, f0);
  mma_step(acc, f1);

  // C/D layout: VGPR r, lanes 0-15 -> M=r, N=lane; lanes 16-31 -> M=r+8, N=lane-16
#pragma unroll
  for (int mi = 0; mi < 2; ++mi) {
#pragma unroll
    for (int r = 0; r < 8; ++r) {
      float* cp = C + (size_t)(mt * 32 + mi * 16 + r + hl * 8) * N + nt * 64 + l16;
      cp[0]  = acc[mi][0][r];
      cp[16] = acc[mi][1][r];
      cp[32] = acc[mi][2][r];
      cp[48] = acc[mi][3][r];
    }
  }
}

// ---------------------------------------------------------------------------
// RoPE on Q: Qf [B*S][H*64] f32 -> Qb [B][H][S][64] bf16
// ---------------------------------------------------------------------------
__global__ void rope_q(const float* __restrict__ Qf, const float* __restrict__ cosT,
                       const float* __restrict__ sinT, bf16_t* __restrict__ Qb,
                       int B, int S) {
  int idx = blockIdx.x * blockDim.x + threadIdx.x;
  if (idx >= B * S * NUM_HEADS * HEAD_DIM) return;
  int d = idx & 63;
  int h = (idx >> 6) & (NUM_HEADS - 1);
  int bs = idx >> 11;          // b*S + s
  int s = bs % S;
  int b = bs / S;

  size_t base = (size_t)bs * (NUM_HEADS * HEAD_DIM) + h * HEAD_DIM;
  float xv = Qf[base + d];
  float other = Qf[base + (d ^ 32)];
  float rot = (d < 32) ? -other : other;
  float o = xv * cosT[s * HEAD_DIM + d] + rot * sinT[s * HEAD_DIM + d];
  Qb[((size_t)(b * NUM_HEADS + h) * S + s) * HEAD_DIM + d] = f2bf(o);
}

// ---------------------------------------------------------------------------
// RoPE on K + pack; V packed transposed.
// Kf/Vf [B*S][G*64] f32 -> Kb [B][G][S][64] bf16, Vt [B][G][64][S] bf16
// ---------------------------------------------------------------------------
__global__ void rope_kv(const float* __restrict__ Kf, const float* __restrict__ Vf,
                        const float* __restrict__ cosT, const float* __restrict__ sinT,
                        bf16_t* __restrict__ Kb, bf16_t* __restrict__ Vt,
                        int B, int S) {
  int idx = blockIdx.x * blockDim.x + threadIdx.x;
  if (idx >= B * S * NUM_KV_GROUPS * HEAD_DIM) return;
  int d = idx & 63;
  int g = (idx >> 6) & (NUM_KV_GROUPS - 1);
  int bs = idx >> 9;           // b*S + s
  int s = bs % S;
  int b = bs / S;

  size_t base = (size_t)bs * (NUM_KV_GROUPS * HEAD_DIM) + g * HEAD_DIM;
  float kv = Kf[base + d];
  float ko = Kf[base + (d ^ 32)];
  float rot = (d < 32) ? -ko : ko;
  float korot = kv * cosT[s * HEAD_DIM + d] + rot * sinT[s * HEAD_DIM + d];
  Kb[((size_t)(b * NUM_KV_GROUPS + g) * S + s) * HEAD_DIM + d] = f2bf(korot);
  Vt[((size_t)(b * NUM_KV_GROUPS + g) * HEAD_DIM + d) * S + s] = f2bf(Vf[base + d]);
}

// ---------------------------------------------------------------------------
// Flash attention (causal, GQA). One wave per (b, h, 16-query block).
// Streams 32 keys/iteration: score WMMAs, then V loads for this block and
// K loads for the next block are issued so their latency hides under the
// softmax VALU (exp + shfl reductions) and LDS P-staging turnaround.
// grid = B*H*(S/16)/8 blocks of 256 threads (8 waves).
// ---------------------------------------------------------------------------
__global__ void __launch_bounds__(256, 1)
flash_attn(const bf16_t* __restrict__ Q, const bf16_t* __restrict__ Km,
           const bf16_t* __restrict__ Vt, bf16_t* __restrict__ Ctx,
           int B, int S) {
  __shared__ bf16_t pbuf[8][16][32];   // per-wave 16x32 P tile (1 KB each)

  const int lane = threadIdx.x & 31;
  const int wave = threadIdx.x >> 5;
  const int hl   = lane >> 4;
  const int l16  = lane & 15;

  const int QB = S / 16;
  int gid = blockIdx.x * 8 + wave;
  int qb = gid % QB;
  int h  = (gid / QB) % NUM_HEADS;
  int b  = gid / (QB * NUM_HEADS);
  int grp = h / GROUP_SIZE;

  const bf16_t* Qh = Q  + ((size_t)(b * NUM_HEADS + h) * S + qb * 16) * HEAD_DIM;
  const bf16_t* Kh = Km + ((size_t)(b * NUM_KV_GROUPS + grp) * S) * HEAD_DIM;
  const bf16_t* Vh = Vt + ((size_t)(b * NUM_KV_GROUPS + grp) * HEAD_DIM) * S;

  // Q A-fragments for d = 0..31 (q0) and d = 32..63 (q1), held for the whole pass
  Frag q0, q1;
  q0.h[0] = *(const v8bf*)(Qh + l16 * HEAD_DIM +  0 + hl * 16);
  q0.h[1] = *(const v8bf*)(Qh + l16 * HEAD_DIM +  8 + hl * 16);
  q1.h[0] = *(const v8bf*)(Qh + l16 * HEAD_DIM + 32 + hl * 16);
  q1.h[1] = *(const v8bf*)(Qh + l16 * HEAD_DIM + 40 + hl * 16);

  v8f c0 = {}, c1 = {}, c2 = {}, c3 = {};
  float mrow[8], lrow[8];
#pragma unroll
  for (int r = 0; r < 8; ++r) { mrow[r] = -3.0e30f; lrow[r] = 0.0f; }

  const float scale = 0.125f;   // 1/sqrt(64)
  const int nkb = qb / 2 + 1;   // key blocks of 32 covering keys <= qb*16+15

  auto loadK = [&](int kbase, v16bf& o00, v16bf& o01, v16bf& o10, v16bf& o11) {
    const bf16_t* k0 = Kh + (size_t)(kbase + l16) * HEAD_DIM + hl * 16;
    o00 = *(const v16bf*)(k0);        // d = hl*16..+15
    o01 = *(const v16bf*)(k0 + 32);   // d = 32 + hl*16..+15
    const bf16_t* k1 = Kh + (size_t)(kbase + 16 + l16) * HEAD_DIM + hl * 16;
    o10 = *(const v16bf*)(k1);
    o11 = *(const v16bf*)(k1 + 32);
  };

  v16bf kb00, kb01, kb10, kb11;
  loadK(0, kb00, kb01, kb10, kb11);

  for (int kb = 0; kb < nkb; ++kb) {
    const int kbase = kb * 32;

    // ----- scores: two 16x16 N-tiles (keys kbase..+15, kbase+16..+31) -----
    v8f s0 = {}, s1 = {};
    s0 = WMMA_BF16(q0.v, kb00, s0);
    s0 = WMMA_BF16(q1.v, kb01, s0);
    s1 = WMMA_BF16(q0.v, kb10, s1);
    s1 = WMMA_BF16(q1.v, kb11, s1);

    // ----- issue V loads (this block) + K loads (next block) early -----
    const bf16_t* vb = Vh + kbase + hl * 16;
    v16bf v0 = *(const v16bf*)(vb + (size_t)( 0 + l16) * S);
    v16bf v1 = *(const v16bf*)(vb + (size_t)(16 + l16) * S);
    v16bf v2 = *(const v16bf*)(vb + (size_t)(32 + l16) * S);
    v16bf v3 = *(const v16bf*)(vb + (size_t)(48 + l16) * S);
    v16bf nk00, nk01, nk10, nk11;
    if (kb + 1 < nkb) loadK(kbase + 32, nk00, nk01, nk10, nk11);

    // ----- scale + causal mask + per-row max -----
    float mx[8];
#pragma unroll
    for (int r = 0; r < 8; ++r) {
      int qrow = qb * 16 + r + hl * 8;
      int key0 = kbase + l16;
      int key1 = kbase + 16 + l16;
      float a0 = s0[r] * scale; if (key0 > qrow) a0 = -3.0e30f;
      float a1 = s1[r] * scale; if (key1 > qrow) a1 = -3.0e30f;
      s0[r] = a0; s1[r] = a1;
      float v = fmaxf(a0, a1);
      v = fmaxf(v, __shfl_xor(v, 1, 32));
      v = fmaxf(v, __shfl_xor(v, 2, 32));
      v = fmaxf(v, __shfl_xor(v, 4, 32));
      v = fmaxf(v, __shfl_xor(v, 8, 32));
      mx[r] = v;
    }

    // ----- online softmax: rescale, exp, store P to LDS as we go -----
    bf16_t (*pw)[32] = pbuf[wave];
    float ps[8];
#pragma unroll
    for (int r = 0; r < 8; ++r) {
      float mnew = fmaxf(mrow[r], mx[r]);
      float corr = __expf(mrow[r] - mnew);
      mrow[r] = mnew;
      lrow[r] *= corr;
      c0[r] *= corr; c1[r] *= corr; c2[r] *= corr; c3[r] *= corr;
      float e0 = __expf(s0[r] - mnew);
      float e1 = __expf(s1[r] - mnew);
      pw[r + hl * 8][l16]      = f2bf(e0);
      pw[r + hl * 8][16 + l16] = f2bf(e1);
      ps[r] = e0 + e1;
    }
    // row-sum reduction (overlaps the outstanding DS stores)
#pragma unroll
    for (int r = 0; r < 8; ++r) {
      float p = ps[r];
      p += __shfl_xor(p, 1, 32);
      p += __shfl_xor(p, 2, 32);
      p += __shfl_xor(p, 4, 32);
      p += __shfl_xor(p, 8, 32);
      lrow[r] += p;
    }

    asm volatile("s_wait_dscnt 0" ::: "memory");   // DS in-order per wave
    Frag pa;
    pa.h[0] = *(const v8bf*)&pw[l16][hl * 16];
    pa.h[1] = *(const v8bf*)&pw[l16][8 + hl * 16];

    // ----- ctx += P(16x32) @ V(32x64) -----
    c0 = WMMA_BF16(pa.v, v0, c0);
    c1 = WMMA_BF16(pa.v, v1, c1);
    c2 = WMMA_BF16(pa.v, v2, c2);
    c3 = WMMA_BF16(pa.v, v3, c3);

    if (kb + 1 < nkb) { kb00 = nk00; kb01 = nk01; kb10 = nk10; kb11 = nk11; }
  }

  // ----- normalize and store ctx as bf16 [B*S][H*64] -----
#pragma unroll
  for (int r = 0; r < 8; ++r) {
    float inv = 1.0f / lrow[r];
    size_t row = (size_t)b * S + qb * 16 + r + hl * 8;
    bf16_t* cp = Ctx + row * (NUM_HEADS * HEAD_DIM) + h * HEAD_DIM + l16;
    cp[0]  = f2bf(c0[r] * inv);
    cp[16] = f2bf(c1[r] * inv);
    cp[32] = f2bf(c2[r] * inv);
    cp[48] = f2bf(c3[r] * inv);
  }
}

// ---------------------------------------------------------------------------
extern "C" void kernel_launch(void* const* d_in, const int* in_sizes, int n_in,
                              void* d_out, int out_size, void* d_ws, size_t ws_size,
                              hipStream_t stream) {
  (void)in_sizes; (void)n_in; (void)out_size; (void)ws_size;

  const float* x    = (const float*)d_in[0];
  const float* Wq   = (const float*)d_in[1];
  const float* Wk   = (const float*)d_in[2];
  const float* Wv   = (const float*)d_in[3];
  const float* Wo   = (const float*)d_in[4];
  const float* cosT = (const float*)d_in[5];
  const float* sinT = (const float*)d_in[6];
  float* out = (float*)d_out;

  const int B = 2, S = 2048, DIN = 2048, DOUT = 2048;
  const int KV = NUM_KV_GROUPS * HEAD_DIM;   // 512
  const int M = B * S;                       // 4096

  uint8_t* ws = (uint8_t*)d_ws;
  size_t off = 0;
  auto alloc = [&](size_t bytes) -> void* {
    void* p = ws + off;
    off += (bytes + 255) & ~(size_t)255;
    return p;
  };
  bf16_t* xb   = (bf16_t*)alloc((size_t)M * DIN * 2);
  bf16_t* wqb  = (bf16_t*)alloc((size_t)DOUT * DIN * 2);
  bf16_t* wkb  = (bf16_t*)alloc((size_t)KV * DIN * 2);
  bf16_t* wvb  = (bf16_t*)alloc((size_t)KV * DIN * 2);
  bf16_t* wob  = (bf16_t*)alloc((size_t)DOUT * DOUT * 2);
  float*  Qf   = (float*)alloc((size_t)M * DOUT * 4);
  float*  Kf   = (float*)alloc((size_t)M * KV * 4);
  float*  Vf   = (float*)alloc((size_t)M * KV * 4);
  bf16_t* Qb   = (bf16_t*)alloc((size_t)M * DOUT * 2);
  bf16_t* Kb   = (bf16_t*)alloc((size_t)M * KV * 2);
  bf16_t* Vtb  = (bf16_t*)alloc((size_t)M * KV * 2);
  bf16_t* ctxb = (bf16_t*)Qf;   // reuse Qf region (dead after rope_q)

  int n4;
  n4 = (M * DIN) / 4;     cvt_f32_bf16_x4<<<(n4 + 255) / 256, 256, 0, stream>>>(x,  xb,  n4);
  n4 = (DOUT * DIN) / 4;  cvt_f32_bf16_x4<<<(n4 + 255) / 256, 256, 0, stream>>>(Wq, wqb, n4);
  n4 = (KV * DIN) / 4;    cvt_f32_bf16_x4<<<(n4 + 255) / 256, 256, 0, stream>>>(Wk, wkb, n4);
  n4 = (KV * DIN) / 4;    cvt_f32_bf16_x4<<<(n4 + 255) / 256, 256, 0, stream>>>(Wv, wvb, n4);
  n4 = (DOUT * DOUT) / 4; cvt_f32_bf16_x4<<<(n4 + 255) / 256, 256, 0, stream>>>(Wo, wob, n4);

  // QKV projections (bf16 WMMA, fp32 accumulate); 32x64 tile per wave
  gemm_bf16_nt<<<dim3(DOUT / 512, M / 32), 256, 0, stream>>>(xb, wqb, Qf, M, DOUT, DIN);
  gemm_bf16_nt<<<dim3(1, M / 32), 256, 0, stream>>>(xb, wkb, Kf, M, KV, DIN);
  gemm_bf16_nt<<<dim3(1, M / 32), 256, 0, stream>>>(xb, wvb, Vf, M, KV, DIN);

  int n;
  n = B * S * NUM_HEADS * HEAD_DIM;
  rope_q<<<(n + 255) / 256, 256, 0, stream>>>(Qf, cosT, sinT, Qb, B, S);
  n = B * S * NUM_KV_GROUPS * HEAD_DIM;
  rope_kv<<<(n + 255) / 256, 256, 0, stream>>>(Kf, Vf, cosT, sinT, Kb, Vtb, B, S);

  // Flash attention: B*H*(S/16) waves, 8 waves per block
  flash_attn<<<(B * NUM_HEADS * (S / 16)) / 8, 256, 0, stream>>>(Qb, Kb, Vtb, ctxb, B, S);

  // Output projection
  gemm_bf16_nt<<<dim3(DOUT / 512, M / 32), 256, 0, stream>>>(ctxb, wob, out, M, DOUT, DOUT);
}
